// MetaSim_56925496541280
// MI455X (gfx1250) — matrix operand order repaired and checked by reading, not compiled, not run on hardware
//
#include <hip/hip_runtime.h>
#include <hip/hip_bf16.h>

typedef __attribute__((ext_vector_type(16))) _Float16 v16h;
typedef __attribute__((ext_vector_type(8)))  _Float16 v8h;
typedef __attribute__((ext_vector_type(4)))  _Float16 v4h;
typedef __attribute__((ext_vector_type(8)))  float    v8f;

#define B_  64
#define N_  1024
#define F_  512
#define P_  32      // padded small-feature width
#define FRAG_LD 24  // halves per fragment row: 16 data + 8 pad => 48B stride (16B aligned)

__device__ __forceinline__ v8f wmma_f16(v16h a, v16h b, v8f c) {
    return __builtin_amdgcn_wmma_f32_16x16x32_f16(
        /*neg_a=*/false, a, /*neg_b=*/false, b,
        /*c_mod=*/(short)0, c, /*reuse_a=*/false, /*reuse_b=*/false);
}

__device__ __forceinline__ v16h frag_load(const _Float16* p) {
    // two aligned 16B LDS loads -> one v16h operand
    const v8h lo = *reinterpret_cast<const v8h*>(p);
    const v8h hi = *reinterpret_cast<const v8h*>(p + 8);
    return __builtin_shufflevector(lo, hi, 0, 1, 2, 3, 4, 5, 6, 7,
                                   8, 9, 10, 11, 12, 13, 14, 15);
}

// A-tile scatter position: row r (0..127), K c (0..31) of the 128x32 tile.
//   wave_tile = r>>4, lane = (r&15) + 16*((c>>3)&1), elem = (c&7) | ((c>>4)<<3)
// B-tile scatter position: K k (0..31), col (0..31):
//   col_tile = col>>4, lane = (col&15) + 16*((k>>3)&1), elem = (k&7) | ((k>>4)<<3)

// ---------------------------------------------------------------------------
// Z[b] = adj[b] (N x N) @ Y[b] (N x 32)  [+ bias, relu]   (f16 WMMA, f32 acc)
// MODE 0: adj read as fp32 (convert on the fly)
// MODE 1: adj read as fp32; converted f16 tiles also written to adjH_out
//         (pass 1 builds a 134MB f16 copy that fits in the 192MB L2)
// MODE 2: adj read from the f16 copy (passes 2-4; L2-resident traffic)
// grid: (N_/128, B_), block 256 (8 waves). Each wave: 16 rows x 32 cols.
// Software pipeline: issue global loads early; frag-load+WMMA before the
// cvt+ds_store of the next tile so the pre-WMMA dscnt wait covers only frags.
// ---------------------------------------------------------------------------
template<int MODE>
__global__ __launch_bounds__(256) void adj_gemm_t(
    const float* __restrict__ adjF, const _Float16* __restrict__ adjH_in,
    _Float16* __restrict__ adjH_out,
    const float* __restrict__ Yin, float* __restrict__ Zout,
    const float* __restrict__ bias, int Ftrue, int relu)
{
    __shared__ __align__(16) _Float16 ldsA[2][8][32][FRAG_LD];  // 24 KB
    __shared__ __align__(16) _Float16 ldsB[2][2][32][FRAG_LD];  //  6 KB

    const int b    = blockIdx.y;
    const int row0 = blockIdx.x * 128;
    const int tid  = threadIdx.x;
    const int wave = tid >> 5;
    const int lane = tid & 31;
    const int m    = lane & 15;
    const int h    = lane >> 4;

    const float*    adjFB = adjF    ? adjF    + (size_t)b * N_ * N_ : nullptr;
    const _Float16* adjHB = adjH_in ? adjH_in + (size_t)b * N_ * N_ : nullptr;
    _Float16*       adjOB = adjH_out? adjH_out+ (size_t)b * N_ * N_ : nullptr;
    const float*    yB    = Yin + (size_t)b * N_ * P_;

    float4 ga[4];   // staged A tile values, fp32 source (MODE 0/1)
    v4h    ha[4];   // staged A tile values, f16 source  (MODE 2)
    float  gb[4];   // staged B tile values

    auto issue_loads = [&](int k0) {
        #pragma unroll
        for (int it = 0; it < 4; ++it) {
            const int i = tid + it * 256;        // 0..1023
            const int r = i >> 3;                // 0..127
            const int c = (i & 7) << 2;          // K base, multiple of 4
            if constexpr (MODE == 2) {
                const _Float16* gp = adjHB + (size_t)(row0 + r) * N_ + (k0 + c);
                ha[it] = *reinterpret_cast<const v4h*>(gp);
                __builtin_prefetch(gp + 32, 0, 3);   // speculative; safe unguarded
            } else {
                const float* gp = adjFB + (size_t)(row0 + r) * N_ + (k0 + c);
                ga[it] = *reinterpret_cast<const float4*>(gp);
                __builtin_prefetch(gp + 32, 0, 3);
            }
        }
        const int col = tid >> 3;                // 0..31
        const int kb  = (tid & 7) << 2;          // K base, multiple of 4
        #pragma unroll
        for (int u = 0; u < 4; ++u)
            gb[u] = yB[(size_t)(k0 + kb + u) * P_ + col];
    };

    auto store_lds = [&](int buf, int k0) {
        #pragma unroll
        for (int it = 0; it < 4; ++it) {
            const int i = tid + it * 256;
            const int r = i >> 3;
            const int c = (i & 7) << 2;
            const int w  = r >> 4, mm = r & 15;
            const int hh = (c >> 3) & 1;
            const int e0 = (c & 7) | ((c >> 4) << 3);
            if constexpr (MODE == 2) {
                *reinterpret_cast<v4h*>(&ldsA[buf][w][mm + 16 * hh][e0]) = ha[it];
            } else {
                const v4h pk = { (_Float16)ga[it].x, (_Float16)ga[it].y,
                                 (_Float16)ga[it].z, (_Float16)ga[it].w };
                if constexpr (MODE == 1)
                    *reinterpret_cast<v4h*>(
                        &adjOB[(size_t)(row0 + r) * N_ + (k0 + c)]) = pk;
                *reinterpret_cast<v4h*>(&ldsA[buf][w][mm + 16 * hh][e0]) = pk;
            }
        }
        {
            const int col = tid >> 3;
            const int kb  = (tid & 7) << 2;
            const v4h pk = { (_Float16)gb[0], (_Float16)gb[1],
                             (_Float16)gb[2], (_Float16)gb[3] };
            const int ct = col >> 4, n = col & 15;
            const int hh = (kb >> 3) & 1;
            const int e0 = (kb & 7) | ((kb >> 4) << 3);
            *reinterpret_cast<v4h*>(&ldsB[buf][ct][n + 16 * hh][e0]) = pk;
        }
    };

    v8f acc0 = {}; v8f acc1 = {};

    const int nstep = N_ / 32;
    issue_loads(0);
    store_lds(0, 0);
    for (int step = 0; step < nstep; ++step) {
        const int cur = step & 1;
        const bool more = (step + 1 < nstep);
        if (more) issue_loads((step + 1) * 32);   // start HBM early
        __syncthreads();                          // tile[step] visible
        const v16h a  = frag_load(&ldsA[cur][wave][lane][0]);
        const v16h b0 = frag_load(&ldsB[cur][0][lane][0]);
        const v16h b1 = frag_load(&ldsB[cur][1][lane][0]);
        acc0 = wmma_f16(a, b0, acc0);
        acc1 = wmma_f16(a, b1, acc1);
        if (more) store_lds(cur ^ 1, (step + 1) * 32);  // cvt+store after WMMA
    }

    const int c0 = m, c1 = 16 + m;
    float bias0 = 0.0f, bias1 = 0.0f;
    if (bias) {
        if (c0 < Ftrue) bias0 = bias[c0];
        if (c1 < Ftrue) bias1 = bias[c1];
    }
    #pragma unroll
    for (int r = 0; r < 8; ++r) {
        const int M = wave * 16 + r + 8 * h;
        float v0 = acc0[r] + bias0;
        float v1 = acc1[r] + bias1;
        if (relu) { v0 = fmaxf(v0, 0.0f); v1 = fmaxf(v1, 0.0f); }
        float* zrow = Zout + (size_t)b * N_ * P_ + (size_t)(row0 + M) * P_;
        zrow[c0] = v0;
        zrow[c1] = v1;
    }
}

// ---------------------------------------------------------------------------
// Out[M x *] = In[M x ldIn] @ W[KW x FW]  [+ bias, relu]
// grid: (65536/128, ceil(FW/32)), block 256. Pads (k>=KW or f>=FW) read as 0.
// ---------------------------------------------------------------------------
__global__ __launch_bounds__(256) void feat_gemm_kernel(
    const float* __restrict__ In, int ldIn, int Kpad,
    const float* __restrict__ W, int KW, int FW,
    float* __restrict__ Out, int ldOut,
    const float* __restrict__ bias, int relu)
{
    __shared__ __align__(16) _Float16 ldsA[2][8][32][FRAG_LD];
    __shared__ __align__(16) _Float16 ldsB[2][2][32][FRAG_LD];

    const int row0 = blockIdx.x * 128;
    const int col0 = blockIdx.y * 32;
    const int tid  = threadIdx.x;
    const int wave = tid >> 5;
    const int lane = tid & 31;
    const int m    = lane & 15;
    const int h    = lane >> 4;

    float4   ga[4];
    _Float16 gb[4];

    auto issue_loads = [&](int k0) {
        #pragma unroll
        for (int it = 0; it < 4; ++it) {
            const int i = tid + it * 256;
            const int r = i >> 3;
            const int c = (i & 7) << 2;
            const float* gp = &In[(size_t)(row0 + r) * ldIn + (k0 + c)];
            ga[it] = *reinterpret_cast<const float4*>(gp);
            __builtin_prefetch(gp + 32, 0, 3);
        }
        const int col = tid >> 3;
        const int kb  = (tid & 7) << 2;
        const int f   = col0 + col;
        #pragma unroll
        for (int u = 0; u < 4; ++u) {
            const int k = k0 + kb + u;
            gb[u] = (k < KW && f < FW) ? (_Float16)W[(size_t)k * FW + f]
                                       : (_Float16)0.0f;
        }
    };

    auto store_lds = [&](int buf) {
        #pragma unroll
        for (int it = 0; it < 4; ++it) {
            const int i = tid + it * 256;
            const int r = i >> 3;
            const int c = (i & 7) << 2;
            const int w  = r >> 4, mm = r & 15;
            const int hh = (c >> 3) & 1;
            const int e0 = (c & 7) | ((c >> 4) << 3);
            const v4h pk = { (_Float16)ga[it].x, (_Float16)ga[it].y,
                             (_Float16)ga[it].z, (_Float16)ga[it].w };
            *reinterpret_cast<v4h*>(&ldsA[buf][w][mm + 16 * hh][e0]) = pk;
        }
        {
            const int col = tid >> 3;
            const int kb  = (tid & 7) << 2;
            const v4h pk = { gb[0], gb[1], gb[2], gb[3] };
            const int ct = col >> 4, n = col & 15;
            const int hh = (kb >> 3) & 1;
            const int e0 = (kb & 7) | ((kb >> 4) << 3);
            *reinterpret_cast<v4h*>(&ldsB[buf][ct][n + 16 * hh][e0]) = pk;
        }
    };

    v8f acc0 = {}; v8f acc1 = {};

    const int nstep = Kpad / 32;
    issue_loads(0);
    store_lds(0);
    for (int step = 0; step < nstep; ++step) {
        const int cur = step & 1;
        const bool more = (step + 1 < nstep);
        if (more) issue_loads((step + 1) * 32);
        __syncthreads();
        const v16h a  = frag_load(&ldsA[cur][wave][lane][0]);
        const v16h b0 = frag_load(&ldsB[cur][0][lane][0]);
        const v16h b1 = frag_load(&ldsB[cur][1][lane][0]);
        acc0 = wmma_f16(a, b0, acc0);
        acc1 = wmma_f16(a, b1, acc1);
        if (more) store_lds(cur ^ 1);
    }

    const int c0 = col0 + m, c1 = col0 + 16 + m;
    float bias0 = 0.0f, bias1 = 0.0f;
    if (bias) {
        if (c0 < FW) bias0 = bias[c0];
        if (c1 < FW) bias1 = bias[c1];
    }
    #pragma unroll
    for (int r = 0; r < 8; ++r) {
        const int M = wave * 16 + r + 8 * h;
        float v0 = acc0[r] + bias0;
        float v1 = acc1[r] + bias1;
        if (relu) { v0 = fmaxf(v0, 0.0f); v1 = fmaxf(v1, 0.0f); }
        float* orow = Out + (size_t)(row0 + M) * ldOut;
        orow[c0] = v0;
        orow[c1] = v1;
    }
}

// ---------------------------------------------------------------------------
// Per-row: softmax(dec[:,:32]) | sigmoid(dec[:,32:]); sampled; log_probs.
// One wave (32 lanes) per row; 8 rows per block.
// ---------------------------------------------------------------------------
__global__ __launch_bounds__(256) void epilogue_kernel(
    const float* __restrict__ dec, const int* __restrict__ masks,
    const float* __restrict__ noise, float* __restrict__ sampled,
    float* __restrict__ log_probs)
{
    const int wave = threadIdx.x >> 5;
    const int lane = threadIdx.x & 31;
    const size_t row = (size_t)blockIdx.x * 8 + wave;
    const int b = (int)(row / N_);

    const float* drow = dec     + row * F_;
    const int*   mrow = masks   + row * F_;
    const float* nrow = noise   + row * F_;
    float*       srow = sampled + row * F_;

    const float LOG_SIGMA    = -3.9120230054281460f; // ln(0.02)
    const float HALF_LOG_2PI =  0.9189385332046727f;
    const float SIGMA        =  0.02f;

    // softmax over first 32 columns: one lane per class
    float v  = drow[lane];
    float mx = v;
    #pragma unroll
    for (int off = 16; off > 0; off >>= 1) mx = fmaxf(mx, __shfl_xor(mx, off, 32));
    float e = __expf(v - mx);
    float sum = e;
    #pragma unroll
    for (int off = 16; off > 0; off >>= 1) sum += __shfl_xor(sum, off, 32);

    float lp = 0.0f;
    {
        const float act = e / sum;
        const float mk  = (float)mrow[lane];
        const float no  = nrow[lane];
        srow[lane] = act + mk * (SIGMA * no);
        lp += mk * (-0.5f * no * no - LOG_SIGMA - HALF_LOG_2PI);
    }
    for (int c = 32 + lane; c < F_; c += 32) {
        const float act = 1.0f / (1.0f + __expf(-drow[c]));
        const float mk  = (float)mrow[c];
        const float no  = nrow[c];
        srow[c] = act + mk * (SIGMA * no);
        lp += mk * (-0.5f * no * no - LOG_SIGMA - HALF_LOG_2PI);
    }
    #pragma unroll
    for (int off = 16; off > 0; off >>= 1) lp += __shfl_xor(lp, off, 32);
    if (lane == 0) atomicAdd(&log_probs[b], lp);
}

__global__ void zero_kernel(float* p, int n) {
    const int i = blockIdx.x * blockDim.x + threadIdx.x;
    if (i < n) p[i] = 0.0f;
}

// ---------------------------------------------------------------------------
extern "C" void kernel_launch(void* const* d_in, const int* in_sizes, int n_in,
                              void* d_out, int out_size, void* d_ws, size_t ws_size,
                              hipStream_t stream) {
    const float* x     = (const float*)d_in[0];
    const float* adj   = (const float*)d_in[1];
    const int*   masks = (const int*)  d_in[2];
    const float* noise = (const float*)d_in[3];
    const float* We1   = (const float*)d_in[4];
    const float* be1   = (const float*)d_in[5];
    const float* We2   = (const float*)d_in[6];
    const float* be2   = (const float*)d_in[7];
    const float* Wd1   = (const float*)d_in[8];
    const float* bd1   = (const float*)d_in[9];
    const float* Wd2   = (const float*)d_in[10];
    const float* bd2   = (const float*)d_in[11];

    float* out       = (float*)d_out;
    float* dec       = out;                                   // [B,N,F]
    float* sampled   = out + (size_t)B_ * N_ * F_;            // [B,N,F]
    float* log_probs = out + (size_t)2 * B_ * N_ * F_;        // [B]

    float* wsY = (float*)d_ws;                                // [B,N,32] fp32
    float* wsH = wsY + (size_t)B_ * N_ * P_;                  // [B,N,32] fp32
    const size_t baseWS  = (size_t)2 * B_ * N_ * P_ * sizeof(float);   // 16 MiB
    const size_t adjHLen = (size_t)B_ * N_ * N_ * sizeof(_Float16);    // 128 MiB
    const bool   useF16  = (ws_size >= baseWS + adjHLen);
    _Float16* adjH = (_Float16*)((char*)d_ws + baseWS);

    const dim3 blk(256);
    const dim3 gFeat(512, 1);       // 65536 rows / 128
    const dim3 gFeatWide(512, 16);  // 512 output cols / 32
    const dim3 gAdj(8, 64);         // 1024 rows / 128, B batches

    // Encoder layer 1 (reassociated): y1 = x @ We1 ; h = relu(adj@y1 + be1)
    feat_gemm_kernel<<<gFeat, blk, 0, stream>>>(x, F_, F_, We1, F_, 30, wsY, P_, nullptr, 0);
    if (useF16) {
        // pass 1 also emits the f16 adj copy (fits in 192MB L2 for passes 2-4)
        adj_gemm_t<1><<<gAdj, blk, 0, stream>>>(adj, nullptr, adjH, wsY, wsH, be1, 30, 1);
        feat_gemm_kernel<<<gFeat, blk, 0, stream>>>(wsH, P_, P_, We2, 30, 18, wsY, P_, nullptr, 0);
        adj_gemm_t<2><<<gAdj, blk, 0, stream>>>(nullptr, adjH, nullptr, wsY, wsH, be2, 18, 0);
        feat_gemm_kernel<<<gFeat, blk, 0, stream>>>(wsH, P_, P_, Wd1, 18, 30, wsY, P_, nullptr, 0);
        adj_gemm_t<2><<<gAdj, blk, 0, stream>>>(nullptr, adjH, nullptr, wsY, wsH, bd1, 30, 1);
        adj_gemm_t<2><<<gAdj, blk, 0, stream>>>(nullptr, adjH, nullptr, wsH, wsY, nullptr, 0, 0);
    } else {
        adj_gemm_t<0><<<gAdj, blk, 0, stream>>>(adj, nullptr, nullptr, wsY, wsH, be1, 30, 1);
        feat_gemm_kernel<<<gFeat, blk, 0, stream>>>(wsH, P_, P_, We2, 30, 18, wsY, P_, nullptr, 0);
        adj_gemm_t<0><<<gAdj, blk, 0, stream>>>(adj, nullptr, nullptr, wsY, wsH, be2, 18, 0);
        feat_gemm_kernel<<<gFeat, blk, 0, stream>>>(wsH, P_, P_, Wd1, 18, 30, wsY, P_, nullptr, 0);
        adj_gemm_t<0><<<gAdj, blk, 0, stream>>>(adj, nullptr, nullptr, wsY, wsH, bd1, 30, 1);
        adj_gemm_t<0><<<gAdj, blk, 0, stream>>>(adj, nullptr, nullptr, wsH, wsY, nullptr, 0, 0);
    }
    // Decoder layer 2 tail: dec = z4 @ Wd2 + bd2
    feat_gemm_kernel<<<gFeatWide, blk, 0, stream>>>(wsY, P_, P_, Wd2, 30, F_, dec, F_, bd2, 0);
    // Activations, sampling, log-probs
    zero_kernel<<<1, 64, 0, stream>>>(log_probs, B_);
    epilogue_kernel<<<(B_ * N_) / 8, blk, 0, stream>>>(dec, masks, noise, sampled, log_probs);
}
